// MultiPathTransformerLayer_18502719111733
// MI455X (gfx1250) — matrix operand order repaired and checked by reading, not compiled, not run on hardware
//
#include <hip/hip_runtime.h>

// ---------------------------------------------------------------------------
// MI455X (gfx1250). Compute-bound layer (~145 GFLOP vs ~100MB HBM at 23.3TB/s)
// -> all GEMMs via v_wmma_f32_16x16x32_bf16. Activations bf16, accumulate fp32.
// * Every LDS tile stored K-contiguous per fragment row -> frag = 2x ds_load_b128
// * global_load_async_to_lds_b128 staging, ping-pong double buffer, ONE barrier
//   per tile (async for tile i+1 overlaps WMMA on tile i)
// * All index math int32; elementwise kernels use grid mapping (no divisions)
// ---------------------------------------------------------------------------

typedef __attribute__((ext_vector_type(16))) __bf16 v16bf;
typedef __attribute__((ext_vector_type(8)))  __bf16 v8bf;
typedef __attribute__((ext_vector_type(8)))  float  v8f;
typedef __attribute__((ext_vector_type(4)))  __bf16 v4bf;

#define DEV __device__ __forceinline__

DEV v8f wmma_bf16(v16bf a, v16bf b, v8f c) {
  return __builtin_amdgcn_wmma_f32_16x16x32_bf16(false, a, false, b, (short)0, c, false, false);
}

// Async global -> LDS, 16 bytes per active lane (VGLOBAL, ASYNCcnt).
DEV void async_g2l_b128(void* lds_ptr, const void* gptr) {
  unsigned int l = (unsigned int)(uintptr_t)lds_ptr;  // low 32 bits = LDS offset
  asm volatile("global_load_async_to_lds_b128 %0, %1, off"
               :: "v"(l), "v"(gptr) : "memory");
}
DEV void wait_async0() { asm volatile("s_wait_asynccnt 0" ::: "memory"); }
DEV void wait_ds0()    { asm volatile("s_wait_dscnt 0" ::: "memory"); }

union FragU { v16bf v; v8bf h[2]; };

// A-layout fragment (16 M x 32 K) from row-major (M, K-contig) tile.
DEV v16bf frag_A(const __bf16* s, int ld) {
  int lane = threadIdx.x & 31, m = lane & 15, h = lane >> 4;
  FragU u;
  u.h[0] = *(const v8bf*)(s + m * ld + h * 8);
  u.h[1] = *(const v8bf*)(s + m * ld + 16 + h * 8);
  return u.v;
}
// B-layout fragment (32 K x 16 N) from (N-row, K-contig) tile.
DEV v16bf frag_B(const __bf16* s, int ld) {
  int lane = threadIdx.x & 31, n = lane & 15, h = lane >> 4;
  FragU u;
  u.h[0] = *(const v8bf*)(s + n * ld + h * 8);
  u.h[1] = *(const v8bf*)(s + n * ld + 16 + h * 8);
  return u.v;
}

DEV float gelu_tanh(float t) {
  return 0.5f * t * (1.0f + tanhf(0.7978845608f * (t + 0.044715f * t * t * t)));
}

// ---------------------------------------------------------------------------
// norm1 -> h^T (N, L, C) bf16. grid = (L, N), block = C/4 = 128. No division.
// ---------------------------------------------------------------------------
__global__ __launch_bounds__(128) void norm_t_kernel(
    const float* __restrict__ x, const float* __restrict__ g,
    const float* __restrict__ b, const float* __restrict__ m,
    const float* __restrict__ v, __bf16* __restrict__ outT, int C, int L) {
  int c = threadIdx.x * 4;
  int l = blockIdx.x, n = blockIdx.y;
  float4 gv = *(const float4*)&g[c];
  float4 bv = *(const float4*)&b[c];
  float4 mv = *(const float4*)&m[c];
  float4 vv = *(const float4*)&v[c];
  const float* xp = x + (n * C + c) * L + l;
  v4bf o;
  { float inv = gv.x * rsqrtf(vv.x + 1e-5f); o[0] = (__bf16)(xp[0]     * inv + (bv.x - mv.x * inv)); }
  { float inv = gv.y * rsqrtf(vv.y + 1e-5f); o[1] = (__bf16)(xp[L]     * inv + (bv.y - mv.y * inv)); }
  { float inv = gv.z * rsqrtf(vv.z + 1e-5f); o[2] = (__bf16)(xp[2 * L] * inv + (bv.z - mv.z * inv)); }
  { float inv = gv.w * rsqrtf(vv.w + 1e-5f); o[3] = (__bf16)(xp[3 * L] * inv + (bv.w - mv.w * inv)); }
  *(v4bf*)&outT[(n * L + l) * C + c] = o;
}

// ---------------------------------------------------------------------------
// pooling (mean+max over r) from h^T -> xa0^T (N, Lk, d).
// grid = (Lk/4, N), block = 128: lanes 0-31*4 = channel, tid>>5 = j sub-index.
// ---------------------------------------------------------------------------
__global__ __launch_bounds__(128) void pool_t_kernel(
    const __bf16* __restrict__ hT, __bf16* __restrict__ outT,
    int chan_off, int C, int L, int r, int Lk, int d) {
  int c = (threadIdx.x & 31) * 4;
  int j = blockIdx.x * 4 + (threadIdx.x >> 5);
  int n = blockIdx.y;
  const __bf16* src = hT + (n * L + j * r) * C + chan_off + c;
  float s0 = 0.f, s1 = 0.f, s2 = 0.f, s3 = 0.f;
  float m0 = -1e30f, m1 = -1e30f, m2 = -1e30f, m3 = -1e30f;
  for (int t = 0; t < r; ++t) {
    v4bf vv = *(const v4bf*)(src + t * C);
    float f0 = (float)vv[0], f1 = (float)vv[1], f2 = (float)vv[2], f3 = (float)vv[3];
    s0 += f0; s1 += f1; s2 += f2; s3 += f3;
    m0 = fmaxf(m0, f0); m1 = fmaxf(m1, f1); m2 = fmaxf(m2, f2); m3 = fmaxf(m3, f3);
  }
  float ir = 1.0f / (float)r;
  v4bf o;
  o[0] = (__bf16)(s0 * ir + m0);
  o[1] = (__bf16)(s1 * ir + m1);
  o[2] = (__bf16)(s2 * ir + m2);
  o[3] = (__bf16)(s3 * ir + m3);
  *(v4bf*)&outT[(n * Lk + j) * d + c] = o;
}

// ---------------------------------------------------------------------------
// Tiled bf16 WMMA GEMM, double-buffered: D(M x Ncols) = A(M x K) * B(K x Ncols).
// B supplied transposed (N, Ncols, K) row stride ldb. Block 128x64, 8 waves of
// 32x32, BK=32. Loop: wait+barrier(tile i) -> issue async(tile i+1) -> 4 WMMA.
// MODE: 0 bias | 1 bias*0.125 | 2 double-BN | 3 bias+gelu |
//       4 bias+resid->fp32 rm AND fused norm2->bf16 T | 5 bias+resid->fp32 rm
// ---------------------------------------------------------------------------
template <int MODE>
__global__ __launch_bounds__(256) void gemm_bf16_kernel(
    const float* __restrict__ A, const __bf16* __restrict__ Bt,
    int K, int Ncols, int ldb, int strideB, int strideD,
    int b_follows_m, int t_store, int Mt,
    const float* __restrict__ bias,
    const float* __restrict__ p0, const float* __restrict__ p1,
    const float* __restrict__ p2, const float* __restrict__ p3,
    const float* __restrict__ p4, const float* __restrict__ p5,
    const float* __restrict__ p6, const float* __restrict__ p7,
    const float* __restrict__ resid,
    float* __restrict__ outf, __bf16* __restrict__ outb) {
  __shared__ __align__(16) __bf16 sA[2][128 * 40];
  __shared__ __align__(16) __bf16 sBt[2][64 * 40];
  const int tid = threadIdx.x, wid = tid >> 5, lane = tid & 31;
  const int n = blockIdx.z;
  const int m0 = blockIdx.y * 128;
  const int c0 = blockIdx.x * 64;
  const float* Ap = A + m0 * K;
  const __bf16* Bp = Bt + n * strideB + c0 * ldb + (b_follows_m ? m0 : 0);
  const int wm = (wid & 3) * 32, wn = (wid >> 2) * 32;

  v8f acc[2][2];
#pragma unroll
  for (int i = 0; i < 2; ++i)
#pragma unroll
    for (int j = 0; j < 2; ++j)
#pragma unroll
      for (int v = 0; v < 8; ++v) acc[i][j][v] = 0.f;

  const int bcol = tid >> 2, bk8 = (tid & 3) * 8;  // B staging (64 cols x 32 k)
  const int ar = tid >> 1, acb = (tid & 1) * 16;   // A staging (128 x 32)
  auto stage = [&](int p, int k0) {
    async_g2l_b128(&sBt[p][bcol * 40 + bk8], Bp + bcol * ldb + k0 + bk8);
    const float* src = Ap + ar * K + k0 + acb;
#pragma unroll
    for (int j = 0; j < 16; ++j) sA[p][ar * 40 + acb + j] = (__bf16)src[j];
  };

  stage(0, 0);
  const int iters = K >> 5;
  for (int i = 0; i < iters; ++i) {
    const int p = i & 1, k0 = i << 5;
    wait_async0();
    __syncthreads();  // tile i visible to all; tile i-1 readers all finished
    if (i + 1 < iters) stage(p ^ 1, k0 + 32);  // overlap next tile with WMMA
    if (i + 2 < iters)
      __builtin_prefetch(Bp + bcol * ldb + k0 + 64 + bk8, 0, 1);
    v16bf a0 = frag_A(&sA[p][(wm) * 40], 40);
    v16bf a1 = frag_A(&sA[p][(wm + 16) * 40], 40);
    v16bf b0 = frag_B(&sBt[p][(wn) * 40], 40);
    v16bf b1 = frag_B(&sBt[p][(wn + 16) * 40], 40);
    acc[0][0] = wmma_bf16(a0, b0, acc[0][0]);
    acc[0][1] = wmma_bf16(a0, b1, acc[0][1]);
    acc[1][0] = wmma_bf16(a1, b0, acc[1][0]);
    acc[1][1] = wmma_bf16(a1, b1, acc[1][1]);
  }

  const int lm = lane & 15, lh = lane >> 4;
#pragma unroll
  for (int tm = 0; tm < 2; ++tm)
#pragma unroll
    for (int tn = 0; tn < 2; ++tn)
#pragma unroll
      for (int v = 0; v < 8; ++v) {
        int row = m0 + wm + tm * 16 + v + 8 * lh;  // output channel
        int col = c0 + wn + tn * 16 + lm;          // spatial position
        int irm = n * strideD + row * Ncols + col;
        int oix = t_store ? (n * strideD + col * Mt + row) : irm;
        float val = acc[tm][tn][v];
        if (MODE != 2) val += bias[row];
        if (MODE == 0) outb[oix] = (__bf16)val;
        if (MODE == 1) outb[oix] = (__bf16)(val * 0.125f);
        if (MODE == 2) {
          float i1 = p0[row] * rsqrtf(p3[row] + 1e-5f);
          float o1 = p1[row] - p2[row] * i1;
          float i2 = p4[row] * rsqrtf(p7[row] + 1e-5f);
          float o2 = p5[row] - p6[row] * i2;
          outb[oix] = (__bf16)((val * i1 + o1) * i2 + o2);
        }
        if (MODE == 3) outb[oix] = (__bf16)gelu_tanh(val);
        if (MODE == 4) {
          val += resid[irm];
          outf[irm] = val;  // attention residual (fp32, row-major)
          float i2 = p0[row] * rsqrtf(p3[row] + 1e-5f);
          float o2 = p1[row] - p2[row] * i2;
          outb[oix] = (__bf16)(val * i2 + o2);  // fused norm2 -> h2^T
        }
        if (MODE == 5) outf[irm] = val + resid[irm];
      }
}

// ---------------------------------------------------------------------------
// Flash attention, one path, double-buffered K/V chunks.
//   q^T (N, L, C)  -> sQt (query, e) : Q A-frags K-contig
//   k^T (N, Lk, d) -> sK  (key, e)   : S-GEMM B-frags K-contig (K-dim = e)
//   v   (N, d, Lk) -> sV  (e, key)   : PV-GEMM B-frags K-contig (K-dim = key)
// Block = 8 waves x 16 queries; one barrier per 32-key chunk.
// ---------------------------------------------------------------------------
__global__ __launch_bounds__(256) void attn_kernel(
    const __bf16* __restrict__ qbT, const __bf16* __restrict__ kbT,
    const __bf16* __restrict__ vb, __bf16* __restrict__ obT,
    int L, int Lk, int C, int d, int q_bstride, int kv_bstride) {
  __shared__ __align__(16) __bf16 sQt[128 * 72];
  __shared__ __align__(16) __bf16 sK[2][32 * 72];
  __shared__ __align__(16) __bf16 sV[2][64 * 40];
  __shared__ __align__(16) __bf16 lds_p[8][16 * 40];
  const int n = blockIdx.z, head = blockIdx.y;
  const int tid = threadIdx.x, wid = tid >> 5, lane = tid & 31;
  const int lm = lane & 15, lh = lane >> 4;
  const __bf16* qp = qbT + n * q_bstride + head * 64;   // [query*C + e]
  const __bf16* kp = kbT + n * kv_bstride + head * 64;  // [key*d + e]
  const __bf16* vp = vb + n * kv_bstride + head * 64 * Lk;  // [e*Lk + key]
  __bf16* op = obT + n * q_bstride + head * 64;         // [query*C + e]
  const int qblk = blockIdx.x * 128;

  const int kr = tid >> 3, ke8 = (tid & 7) * 8;  // sK staging (32 key x 64 e)
  const int ve = tid >> 2, vk8 = (tid & 3) * 8;  // sV staging (64 e x 32 key)
  auto stage_kv = [&](int p, int m0) {
    async_g2l_b128(&sK[p][kr * 72 + ke8], kp + (m0 + kr) * d + ke8);
    async_g2l_b128(&sV[p][ve * 40 + vk8], vp + ve * Lk + m0 + vk8);
  };

  // stage Q panel (128 q x 64 e, 4 b128/thread) together with first K/V chunk
  {
    int qr = tid >> 1, eb = (tid & 1) * 32;
#pragma unroll
    for (int u = 0; u < 4; ++u)
      async_g2l_b128(&sQt[qr * 72 + eb + u * 8],
                     qp + (qblk + qr) * C + eb + u * 8);
  }
  stage_kv(0, 0);
  wait_async0();
  __syncthreads();

  v16bf aq0 = frag_A(&sQt[(wid * 16) * 72], 72);       // e 0..31
  v16bf aq1 = frag_A(&sQt[(wid * 16) * 72 + 32], 72);  // e 32..63

  v8f o_acc[4];
#pragma unroll
  for (int t = 0; t < 4; ++t)
#pragma unroll
    for (int v = 0; v < 8; ++v) o_acc[t][v] = 0.f;
  float rm[8], rs[8];
#pragma unroll
  for (int v = 0; v < 8; ++v) { rm[v] = -1e30f; rs[v] = 0.f; }

  const int iters = Lk >> 5;
  for (int it = 0; it < iters; ++it) {
    const int p = it & 1;
    if (it > 0) {
      wait_async0();
      __syncthreads();
    }
    if (it + 1 < iters) stage_kv(p ^ 1, (it + 1) << 5);  // overlap with WMMA

    // S = Q^T K
    v16bf bk00 = frag_B(&sK[p][0 * 72 + 0], 72);
    v16bf bk01 = frag_B(&sK[p][16 * 72 + 0], 72);
    v16bf bk10 = frag_B(&sK[p][0 * 72 + 32], 72);
    v16bf bk11 = frag_B(&sK[p][16 * 72 + 32], 72);
    v8f s0, s1;
#pragma unroll
    for (int v = 0; v < 8; ++v) { s0[v] = 0.f; s1[v] = 0.f; }
    s0 = wmma_bf16(aq0, bk00, s0);
    s0 = wmma_bf16(aq1, bk10, s0);
    s1 = wmma_bf16(aq0, bk01, s1);
    s1 = wmma_bf16(aq1, bk11, s1);

    // online softmax (row = v + 8*lh; its 16 lanes reduced via shfl_xor w16)
#pragma unroll
    for (int v = 0; v < 8; ++v) {
      float t = fmaxf(s0[v], s1[v]);
      t = fmaxf(t, __shfl_xor(t, 1, 16));
      t = fmaxf(t, __shfl_xor(t, 2, 16));
      t = fmaxf(t, __shfl_xor(t, 4, 16));
      t = fmaxf(t, __shfl_xor(t, 8, 16));
      float mnew = fmaxf(rm[v], t);
      float sc2 = __expf(rm[v] - mnew);
      rm[v] = mnew;
      float pl = __expf(s0[v] - mnew);
      float ph = __expf(s1[v] - mnew);
      float su = pl + ph;
      su += __shfl_xor(su, 1, 16);
      su += __shfl_xor(su, 2, 16);
      su += __shfl_xor(su, 4, 16);
      su += __shfl_xor(su, 8, 16);
      rs[v] = rs[v] * sc2 + su;
#pragma unroll
      for (int t2 = 0; t2 < 4; ++t2) o_acc[t2][v] *= sc2;
      int prow = v + 8 * lh;
      lds_p[wid][prow * 40 + lm] = (__bf16)pl;
      lds_p[wid][prow * 40 + 16 + lm] = (__bf16)ph;
    }
    wait_ds0();  // wave-private LDS transpose; DS in-order per wave

    v16bf pa = frag_A(&lds_p[wid][0], 40);
#pragma unroll
    for (int t2 = 0; t2 < 4; ++t2) {
      v16bf bvf = frag_B(&sV[p][(t2 * 16) * 40], 40);
      o_acc[t2] = wmma_bf16(pa, bvf, o_acc[t2]);
    }
  }

  // normalize and store O transposed: op[query*C + e]
#pragma unroll
  for (int v = 0; v < 8; ++v) {
    float inv = 1.0f / rs[v];
    int qrow = qblk + wid * 16 + v + 8 * lh;
#pragma unroll
    for (int t2 = 0; t2 < 4; ++t2)
      op[qrow * C + t2 * 16 + lm] = (__bf16)(o_acc[t2][v] * inv);
  }
}

// ---------------------------------------------------------------------------
// host launcher
// ---------------------------------------------------------------------------
extern "C" void kernel_launch(void* const* d_in, const int* in_sizes, int n_in,
                              void* d_out, int out_size, void* d_ws, size_t ws_size,
                              hipStream_t stream) {
  (void)in_sizes; (void)n_in; (void)out_size; (void)ws_size;
  const float* x   = (const float*)d_in[0];
  const float* n1g = (const float*)d_in[1];
  const float* n1b = (const float*)d_in[2];
  const float* n1m = (const float*)d_in[3];
  const float* n1v = (const float*)d_in[4];
  const float* n2g = (const float*)d_in[5];
  const float* n2b = (const float*)d_in[6];
  const float* n2m = (const float*)d_in[7];
  const float* n2v = (const float*)d_in[8];
  const float* Wq  = (const float*)d_in[9];
  const float* Wk  = (const float*)d_in[10];
  const float* Wv  = (const float*)d_in[11];
  const float* Wo  = (const float*)d_in[12];
  const float* Wp  = (const float*)d_in[13];
  const float* bq  = (const float*)d_in[14];
  const float* bk  = (const float*)d_in[15];
  const float* bv  = (const float*)d_in[16];
  const float* bo  = (const float*)d_in[17];
  const float* agg = (const float*)d_in[18];
  const float* agb = (const float*)d_in[19];
  const float* agm = (const float*)d_in[20];
  const float* agv = (const float*)d_in[21];
  const float* ang = (const float*)d_in[22];
  const float* anb = (const float*)d_in[23];
  const float* anm = (const float*)d_in[24];
  const float* anv = (const float*)d_in[25];
  const float* W1  = (const float*)d_in[26];
  const float* b1  = (const float*)d_in[27];
  const float* W2  = (const float*)d_in[28];
  const float* b2  = (const float*)d_in[29];

  const int N = 8, C = 512, L = 3072, d = 128, PATHS = 4, FF = 2048;
  const int ratios[4] = {2, 4, 8, 16};
  const int CL = C * L;
  const long NCL = (long)N * CL;

  // workspace carving (256B aligned)
  char* ws = (char*)d_ws;
  size_t off = 0;
  auto carve = [&](size_t bytes) -> void* {
    off = (off + 255) & ~(size_t)255;
    void* p = ws + off;
    off += bytes;
    return p;
  };
  __bf16* hT   = (__bf16*)carve(NCL * 2);      // norm1(x)^T   (N, L, C)
  __bf16* qT   = (__bf16*)carve(NCL * 2);      // Q^T          (N, L, C)
  __bf16* oT   = (__bf16*)carve(NCL * 2);      // attn out^T   (N, L, C)
  __bf16* h2T  = (__bf16*)carve(NCL * 2);      // norm2(y)^T   (N, L, C)
  float*  ybuf = (float*) carve(NCL * 4);      // x + attn, row-major (N, C, L)
  __bf16* mT   = (__bf16*)carve(NCL * 4 * 2);  // gelu(W1 h2)^T (N, L, FF)
  long pe[5];
  pe[0] = 0;
  for (int i = 0; i < 4; ++i) pe[i + 1] = pe[i] + (long)N * d * (L / ratios[i]);
  __bf16* xa0T = (__bf16*)carve(pe[4] * 2);    // pooled^T (N, Lk, d) per path
  __bf16* xaT  = (__bf16*)carve(pe[4] * 2);    // bn(Wp @ pooled)^T (N, Lk, d)
  __bf16* kT   = (__bf16*)carve(pe[4] * 2);    // K^T (N, Lk, d)
  __bf16* vbf  = (__bf16*)carve(pe[4] * 2);    // V   (N, d, Lk)

  // 1) norm1 -> h^T
  norm_t_kernel<<<dim3(L, N), 128, 0, stream>>>(x, n1g, n1b, n1m, n1v, hT, C, L);

  // 2) pooling per path -> xa0^T
  for (int i = 0; i < PATHS; ++i) {
    int Lk = L / ratios[i];
    pool_t_kernel<<<dim3(Lk / 4, N), 128, 0, stream>>>(
        hT, xa0T + pe[i], i * d, C, L, ratios[i], Lk, d);
  }

  // 3) per path: Wp + double BN -> xa^T ; Wk -> K^T ; Wv -> V (row-major)
  for (int i = 0; i < PATHS; ++i) {
    int Lk = L / ratios[i];
    int sKV = d * Lk;
    gemm_bf16_kernel<2><<<dim3(Lk / 64, 1, N), 256, 0, stream>>>(
        Wp + i * d * d, xa0T + pe[i], d, Lk, d, sKV, sKV, 0, 1, d,
        nullptr,
        agg + i * d, agb + i * d, agm + i * d, agv + i * d,
        ang + i * d, anb + i * d, anm + i * d, anv + i * d,
        nullptr, nullptr, xaT + pe[i]);
    gemm_bf16_kernel<0><<<dim3(Lk / 64, 1, N), 256, 0, stream>>>(
        Wk + i * d * d, xaT + pe[i], d, Lk, d, sKV, sKV, 0, 1, d,
        bk + i * d,
        nullptr, nullptr, nullptr, nullptr, nullptr, nullptr, nullptr, nullptr,
        nullptr, nullptr, kT + pe[i]);
    gemm_bf16_kernel<0><<<dim3(Lk / 64, 1, N), 256, 0, stream>>>(
        Wv + i * d * d, xaT + pe[i], d, Lk, d, sKV, sKV, 0, 0, d,
        bv + i * d,
        nullptr, nullptr, nullptr, nullptr, nullptr, nullptr, nullptr, nullptr,
        nullptr, nullptr, vbf + pe[i]);
  }

  // 4) Q projection (block-diagonal paths via grid.y), 1/sqrt(64) folded
  gemm_bf16_kernel<1><<<dim3(L / 64, PATHS, N), 256, 0, stream>>>(
      Wq, hT, d, L, C, CL, CL, 1, 1, C, bq,
      nullptr, nullptr, nullptr, nullptr, nullptr, nullptr, nullptr, nullptr,
      nullptr, nullptr, qT);

  // 5) flash attention per path (2 heads)
  for (int i = 0; i < PATHS; ++i) {
    int Lk = L / ratios[i];
    attn_kernel<<<dim3(L / 128, 2, N), 256, 0, stream>>>(
        qT + i * d, kT + pe[i], vbf + pe[i], oT + i * d,
        L, Lk, C, d, CL, d * Lk);
  }

  // 6) Wo + residual -> ybuf (fp32 rm) + fused norm2 -> h2^T
  gemm_bf16_kernel<4><<<dim3(L / 64, PATHS, N), 256, 0, stream>>>(
      Wo, oT, d, L, C, CL, CL, 1, 1, C, bo,
      n2g, n2b, n2m, n2v, nullptr, nullptr, nullptr, nullptr,
      x, ybuf, h2T);

  // 7) W1 + gelu: (2048 x 512) x (512 x L) -> m^T
  gemm_bf16_kernel<3><<<dim3(L / 64, FF / 128, N), 256, 0, stream>>>(
      W1, h2T, C, L, C, CL, FF * L, 0, 1, FF, b1,
      nullptr, nullptr, nullptr, nullptr, nullptr, nullptr, nullptr, nullptr,
      nullptr, nullptr, mT);

  // 8) W2 + residual: (512 x 2048) x (2048 x L) -> d_out (fp32 row-major)
  gemm_bf16_kernel<5><<<dim3(L / 64, C / 128, N), 256, 0, stream>>>(
      W2, mT, FF, L, FF, FF * L, CL, 0, 0, C, b2,
      nullptr, nullptr, nullptr, nullptr, nullptr, nullptr, nullptr, nullptr,
      ybuf, (float*)d_out, nullptr);
}
